// MOERounter_64123861729521
// MI455X (gfx1250) — matrix operand order repaired and checked by reading, not compiled, not run on hardware
//
#include <hip/hip_runtime.h>

typedef __bf16 bf16_t;
typedef bf16_t v8bf  __attribute__((ext_vector_type(8)));
typedef bf16_t v16bf __attribute__((ext_vector_type(16)));
typedef float  v4f   __attribute__((ext_vector_type(4)));
typedef float  v8f   __attribute__((ext_vector_type(8)));

#define T_TOK   8192
#define HDIM    4096
#define ENUM    64
#define KC      64          // K-chunk staged in LDS
#define KCP     72          // padded bf16 row stride: 144B rows, 16B aligned
#define TOK_WG  128         // tokens per workgroup (8 waves x 16 tokens)

// ---------------------------------------------------------------------------
// Router GEMM via bf16x3 fp32-emulation on V_WMMA_F32_16X16X32_BF16:
//   x = xh + xl, w = wh + wl (bf16 splits, f32 accumulate)
//   logits ~= xh*wh + xh*wl + xl*wh      (~1e-5 rel accuracy, fp32 accum)
// Each wave: 16 tokens x 64 experts (4 N-tiles). Weights staged as hi/lo
// bf16 planes in LDS, converted once per workgroup.
// ---------------------------------------------------------------------------
__global__ __launch_bounds__(256)
void moe_router_kernel(const float* __restrict__ x,
                       const float* __restrict__ gate_w,
                       const float* __restrict__ gate_b,
                       float* __restrict__ out_logits,
                       float* __restrict__ out_weights,
                       int*   __restrict__ out_sel)
{
    __shared__ bf16_t sWh[ENUM * KCP];      //  9,216 B  weight hi plane
    __shared__ bf16_t sWl[ENUM * KCP];      //  9,216 B  weight lo plane
    __shared__ float  sLog[TOK_WG * ENUM];  // 32,768 B  logits for this WG

    const int tid  = threadIdx.x;
    const int lane = tid & 31;
    const int wave = tid >> 5;
    const int half = lane >> 4;             // K half-group selector
    const int n16  = lane & 15;             // M (A rows) / N (B cols) in tile

    const int tok0 = blockIdx.x * TOK_WG;
    const int mtok = tok0 + wave * 16 + n16;          // this lane's A row
    const float* aptr = x + (size_t)mtok * HDIM;

    v8f acc[4] = {};                                   // 4 x (16x16 f32) tiles

    for (int k0 = 0; k0 < HDIM; k0 += KC) {
        __syncthreads();   // previous slab fully consumed before overwrite
        // ---- stage gate_w[:, k0:k0+KC]: f32 -> bf16 hi/lo planes in LDS ----
        #pragma unroll
        for (int i = 0; i < (ENUM * KC) / (256 * 4); ++i) {
            int f   = i * 256 + tid;                   // float4 index
            int row = f >> 4;                          // 16 float4 per expert
            int col = (f & 15) << 2;
            v4f v = *(const v4f*)(gate_w + (size_t)row * HDIM + k0 + col);
            bf16_t* ph = &sWh[row * KCP + col];
            bf16_t* pl = &sWl[row * KCP + col];
            #pragma unroll
            for (int j = 0; j < 4; ++j) {
                float  fv = v[j];
                bf16_t h  = (bf16_t)fv;
                ph[j] = h;
                pl[j] = (bf16_t)(fv - (float)h);       // residual, exact in f32
            }
        }
        __syncthreads();

        if (k0 + KC < HDIM)
            __builtin_prefetch(aptr + k0 + KC, 0, 1);  // global_prefetch_b8

        #pragma unroll
        for (int kk = 0; kk < KC; kk += 32) {
            // ---- A fragment (16-bit A 16x32 layout):
            // lanes 0-15: K = kk+0..7 and kk+16..23 ; lanes 16-31: +8
            float av[16];
            *(v4f*)&av[0]  = *(const v4f*)(aptr + k0 + kk + half * 8);
            *(v4f*)&av[4]  = *(const v4f*)(aptr + k0 + kk + half * 8 + 4);
            *(v4f*)&av[8]  = *(const v4f*)(aptr + k0 + kk + 16 + half * 8);
            *(v4f*)&av[12] = *(const v4f*)(aptr + k0 + kk + 16 + half * 8 + 4);
            v16bf ah, al;
            #pragma unroll
            for (int j = 0; j < 16; ++j) {
                bf16_t h = (bf16_t)av[j];
                ah[j] = h;
                al[j] = (bf16_t)(av[j] - (float)h);
            }

            // ---- B fragments + 3-term WMMA accumulation per N-tile ----
            const int boff = kk + half * 8;            // lane's K: +0..7 and +16..23
            #pragma unroll
            for (int c = 0; c < 4; ++c) {
                const int ridx = (c * 16 + n16) * KCP + boff;
                v8bf bh0 = *(const v8bf*)(&sWh[ridx]);
                v8bf bh1 = *(const v8bf*)(&sWh[ridx + 16]);
                v8bf bl0 = *(const v8bf*)(&sWl[ridx]);
                v8bf bl1 = *(const v8bf*)(&sWl[ridx + 16]);
                v16bf bh = __builtin_shufflevector(bh0, bh1,
                    0,1,2,3,4,5,6,7,8,9,10,11,12,13,14,15);
                v16bf bl = __builtin_shufflevector(bl0, bl1,
                    0,1,2,3,4,5,6,7,8,9,10,11,12,13,14,15);
                acc[c] = __builtin_amdgcn_wmma_f32_16x16x32_bf16(
                    false, ah, false, bh, (short)0, acc[c], false, false);
                acc[c] = __builtin_amdgcn_wmma_f32_16x16x32_bf16(
                    false, ah, false, bl, (short)0, acc[c], false, false);
                acc[c] = __builtin_amdgcn_wmma_f32_16x16x32_bf16(
                    false, al, false, bh, (short)0, acc[c], false, false);
            }
        }
    }

    // ---- bias + spill accumulators to LDS in [local_token][expert] order ----
    float bc[4];
    #pragma unroll
    for (int c = 0; c < 4; ++c) bc[c] = gate_b[c * 16 + n16];

    #pragma unroll
    for (int c = 0; c < 4; ++c) {
        #pragma unroll
        for (int r = 0; r < 8; ++r) {
            int m = r + 8 * half;                     // C/D layout: M = r (+8 hi)
            sLog[(wave * 16 + m) * ENUM + c * 16 + n16] = acc[c][r] + bc[c];
        }
    }
    __syncthreads();

    // ---- logits to global, coalesced float4 ----
    {
        v4f*       dst = (v4f*)(out_logits + (size_t)tok0 * ENUM);
        const v4f* src = (const v4f*)sLog;
        #pragma unroll
        for (int i = 0; i < (TOK_WG * ENUM) / (256 * 4); ++i)
            dst[i * 256 + tid] = src[i * 256 + tid];
    }

    // ---- per-token top-2 + renormalized softmax (threads 0..127) ----
    if (tid < TOK_WG) {
        const float* l = &sLog[tid * ENUM];
        float m1 = -3.402823466e+38f, m2 = -3.402823466e+38f;
        int   i1 = 0, i2 = 0;
        #pragma unroll
        for (int e = 0; e < ENUM; ++e) {
            float v = l[e];
            if (v > m1)      { m2 = m1; i2 = i1; m1 = v; i1 = e; }
            else if (v > m2) { m2 = v;  i2 = e; }
        }
        // renormalized top-2 softmax == softmax over {m1, m2}
        float e2 = __expf(m2 - m1);
        float w1 = 1.0f / (1.0f + e2);
        float w2 = e2 * w1;
        int t = tok0 + tid;
        out_weights[t * 2 + 0] = w1;
        out_weights[t * 2 + 1] = w2;
        out_sel[t * 2 + 0] = i1;
        out_sel[t * 2 + 1] = i2;
    }
}

// ---------------------------------------------------------------------------
// One-hot expert mask: mask[e][k][t] = (sel[t][k] == e), t fastest -> coalesced
// ---------------------------------------------------------------------------
__global__ __launch_bounds__(256)
void moe_mask_kernel(const int* __restrict__ sel, int* __restrict__ mask)
{
    int f   = blockIdx.x * 256 + threadIdx.x;   // flat over E*TOPK*T
    int t   = f & (T_TOK - 1);
    int rem = f >> 13;                          // T_TOK == 2^13
    int k   = rem & 1;
    int e   = rem >> 1;
    mask[f] = (sel[t * 2 + k] == e) ? 1 : 0;
}

extern "C" void kernel_launch(void* const* d_in, const int* in_sizes, int n_in,
                              void* d_out, int out_size, void* d_ws, size_t ws_size,
                              hipStream_t stream)
{
    (void)in_sizes; (void)n_in; (void)out_size; (void)d_ws; (void)ws_size;

    const float* x      = (const float*)d_in[0];
    const float* gate_w = (const float*)d_in[1];
    const float* gate_b = (const float*)d_in[2];

    // d_out: logits f32 [8192*64] | weights f32 [8192*2] | sel i32 [8192*2] | mask i32 [64*2*8192]
    float* out_logits  = (float*)d_out;
    float* out_weights = out_logits + (size_t)T_TOK * ENUM;              // 524288
    int*   out_sel     = (int*)(out_weights + (size_t)T_TOK * 2);        // 540672
    int*   out_mask    = out_sel + (size_t)T_TOK * 2;                    // 557056

    moe_router_kernel<<<T_TOK / TOK_WG, 256, 0, stream>>>(
        x, gate_w, gate_b, out_logits, out_weights, out_sel);

    moe_mask_kernel<<<(ENUM * 2 * T_TOK) / 256, 256, 0, stream>>>(
        out_sel, out_mask);
}
